// RelationalGraphAttentionConv_30193620091222
// MI455X (gfx1250) — compile-verified, matrix operand order checked by reading, not run on hardware
//
#include <hip/hip_runtime.h>
#include <hip/hip_bf16.h>

// ---------------------------------------------------------------------------
// RGAT conv for MI455X (gfx1250, wave32).
//  hidden[r] = x @ W[r]^T  -> bf16 WMMA (v_wmma_f32_16x16x32_bf16), fp32 accum
//  hidden (115 MB fp32) lives in workspace and fits the 192 MB L2, so the
//  per-edge random gathers (the dominant ~435 MB of traffic) hit L2.
//  Segment softmax via native float atomics + uint-encoded atomic float max.
// ---------------------------------------------------------------------------

typedef __attribute__((ext_vector_type(16))) __bf16 v16bf;
typedef __attribute__((ext_vector_type(8)))  float  v8f;

#define NEG_SLOPE 0.2f
#define EPSV      1e-10f
#define NHEAD     8
#define RP1       9      // R + 1 (self-loop relation id = 8)

union Frag16 { v16bf v; uint4 q[2]; };

// Monotonic mapping float -> uint so unsigned atomicMax == float max.
__device__ __forceinline__ unsigned fkey(float f) {
  unsigned u = __float_as_uint(f);
  return (u & 0x80000000u) ? ~u : (u | 0x80000000u);
}
__device__ __forceinline__ float unfkey(unsigned k) {
  unsigned u = (k & 0x80000000u) ? (k & 0x7FFFFFFFu) : ~k;
  return __uint_as_float(u);
}

__device__ __forceinline__ void atomAddF(float* p, float v) {
  unsafeAtomicAdd(p, v);   // native global_atomic_add_f32 on CDNA
}

// ---- convert x and W_tau to bf16 ------------------------------------------
__global__ void k_prep_bf16(const float* __restrict__ x, const float* __restrict__ W,
                            __bf16* __restrict__ xh, __bf16* __restrict__ Wh,
                            int nx, int nw) {
  int i = blockIdx.x * blockDim.x + threadIdx.x;
  if (i < nx) xh[i] = (__bf16)x[i];
  if (i < nw) Wh[i] = (__bf16)W[i];
}

__global__ void k_zero_u32(unsigned* __restrict__ p, int n) {
  int i = blockIdx.x * blockDim.x + threadIdx.x;
  if (i < n) p[i] = 0u;
}

// ---- hidden[r,n,d] = sum_k x[n,k] * W[r,d,k], bf16 WMMA -------------------
// One wave computes a 16-row x 64-col strip for one relation:
//   4 column tiles x (2 WMMA K-steps of 32) = 8 v_wmma per wave.
__global__ void k_hidden_wmma(const __bf16* __restrict__ xh, const __bf16* __restrict__ Wh,
                              float* __restrict__ hidden, int N) {
  const int nrb   = N >> 4;              // N % 16 == 0 (50000 = 3125*16)
  const int ntask = RP1 * nrb;
  const int wave  = blockIdx.x * (blockDim.x >> 5) + (threadIdx.x >> 5);
  if (wave >= ntask) return;             // wave-uniform: EXEC stays all-1s for WMMA
  const int r    = wave / nrb;
  const int row0 = (wave - r * nrb) << 4;
  const int lane = threadIdx.x & 31;
  const int half = lane >> 4;            // K half select per 16-bit A/B layout
  const int m    = lane & 15;            // row within tile (A), col within tile (B)

  // A fragments (x rows): lane holds row m, K in {8h..8h+7} U {16+8h..16+8h+7}
  Frag16 A0, A1;
  const char* xb = (const char*)(xh + (size_t)(row0 + m) * 64);
  A0.q[0] = *(const uint4*)(xb + 16 * half);
  A0.q[1] = *(const uint4*)(xb + 32 + 16 * half);
  A1.q[0] = *(const uint4*)(xb + 64 + 16 * half);
  A1.q[1] = *(const uint4*)(xb + 96 + 16 * half);

  #pragma unroll
  for (int t = 0; t < 4; ++t) {
    const int dcol0 = t << 4;
    // B fragments: column d = W[r][dcol0+d][.] is contiguous in K.
    Frag16 B0, B1;
    const char* wb = (const char*)(Wh + ((size_t)r * 64 + dcol0 + m) * 64);
    B0.q[0] = *(const uint4*)(wb + 16 * half);
    B0.q[1] = *(const uint4*)(wb + 32 + 16 * half);
    B1.q[0] = *(const uint4*)(wb + 64 + 16 * half);
    B1.q[1] = *(const uint4*)(wb + 96 + 16 * half);

    v8f c = {};
    c = __builtin_amdgcn_wmma_f32_16x16x32_bf16(false, A0.v, false, B0.v,
                                                (short)0, c, false, false);
    c = __builtin_amdgcn_wmma_f32_16x16x32_bf16(false, A1.v, false, B1.v,
                                                (short)0, c, false, false);

    // D layout: VGPR q -> row (q + 8*half), col = lane&15
    float* op = hidden + ((size_t)r * N + row0 + 8 * half) * 64 + dcol0 + m;
    #pragma unroll
    for (int qi = 0; qi < 8; ++qi) op[(size_t)qi * 64] = c[qi];
  }
}

// ---- pass 1: attention logits + segment max + degree count ----------------
// one thread per (edge, head): gathers 2 x 32B, 16 FMAs.
__global__ void k_edge_logits(const float* __restrict__ hidden, const float* __restrict__ query,
                              const int* __restrict__ node_in, const int* __restrict__ node_out,
                              const int* __restrict__ relation,
                              float* __restrict__ wbuf, unsigned* __restrict__ mxkey,
                              int* __restrict__ cnt, int N, int E, int M) {
  int gid = blockIdx.x * blockDim.x + threadIdx.x;
  if (gid >= M * NHEAD) return;
  int m = gid >> 3, h = gid & 7;
  int ni, no, rel;
  if (m < E) { ni = node_in[m]; no = node_out[m]; rel = relation[m]; }
  else       { ni = m - E; no = ni; rel = RP1 - 1; }           // self-loop

  const float4* hi4 = (const float4*)(hidden + ((size_t)rel * N + ni) * 64 + h * 8);
  const float4* ho4 = (const float4*)(hidden + ((size_t)rel * N + no) * 64 + h * 8);
  const float4* q4  = (const float4*)(query + ((size_t)rel * NHEAD + h) * 16);
  float4 A0 = hi4[0], A1 = hi4[1];
  float4 B0 = ho4[0], B1 = ho4[1];
  float4 Q0 = q4[0], Q1 = q4[1], Q2 = q4[2], Q3 = q4[3];

  // key_feat interleaves (h_in, h_out): w = sum_k q[2k]*hi[k] + q[2k+1]*ho[k]
  float w = Q0.x*A0.x + Q0.y*B0.x + Q0.z*A0.y + Q0.w*B0.y
          + Q1.x*A0.z + Q1.y*B0.z + Q1.z*A0.w + Q1.w*B0.w
          + Q2.x*A1.x + Q2.y*B1.x + Q2.z*A1.y + Q2.w*B1.y
          + Q3.x*A1.z + Q3.y*B1.z + Q3.z*A1.w + Q3.w*B1.w;
  w = (w > 0.f) ? w : NEG_SLOPE * w;                            // leaky relu
  wbuf[gid] = w;
  atomicMax(&mxkey[no * NHEAD + h], fkey(w));
  if (h == 0) atomicAdd(&cnt[no], 1);
}

// ---- pass 2: exp(w - max) * edge_weight, segment sum ----------------------
__global__ void k_edge_exp(const int* __restrict__ node_out, const float* __restrict__ edge_weight,
                           float* __restrict__ wbuf, const unsigned* __restrict__ mxkey,
                           float* __restrict__ ssum, int E, int M) {
  int gid = blockIdx.x * blockDim.x + threadIdx.x;
  if (gid >= M * NHEAD) return;
  int m = gid >> 3, h = gid & 7;
  int no; float ew;
  if (m < E) { no = node_out[m]; ew = edge_weight[m]; }
  else       { no = m - E; ew = 1.f; }
  float mx = unfkey(mxkey[no * NHEAD + h]);
  float a = __expf(wbuf[gid] - mx) * ew;
  wbuf[gid] = a;                                                // reuse buffer as att
  atomAddF(&ssum[no * NHEAD + h], a);
}

// ---- pass 3: normalize (scatter-mean denominator) + weighted scatter ------
__global__ void k_edge_scatter(const float* __restrict__ hidden,
                               const int* __restrict__ node_in, const int* __restrict__ node_out,
                               const int* __restrict__ relation,
                               const float* __restrict__ wbuf, const float* __restrict__ ssum,
                               const int* __restrict__ cnt, float* __restrict__ accum,
                               int N, int E, int M) {
  int gid = blockIdx.x * blockDim.x + threadIdx.x;
  if (gid >= M * NHEAD) return;
  int m = gid >> 3, h = gid & 7;
  int ni, no, rel;
  if (m < E) { ni = node_in[m]; no = node_out[m]; rel = relation[m]; }
  else       { ni = m - E; no = ni; rel = RP1 - 1; }
  float a    = wbuf[gid];
  float c    = (float)cnt[no];
  float norm = ssum[no * NHEAD + h] / c;                        // scatter-MEAN normalizer
  float att  = a / (norm + EPSV);
  const float4* hi4 = (const float4*)(hidden + ((size_t)rel * N + ni) * 64 + h * 8);
  float4 A0 = hi4[0], A1 = hi4[1];
  float* ap = accum + (size_t)no * 64 + h * 8;
  atomAddF(ap + 0, att * A0.x); atomAddF(ap + 1, att * A0.y);
  atomAddF(ap + 2, att * A0.z); atomAddF(ap + 3, att * A0.w);
  atomAddF(ap + 4, att * A1.x); atomAddF(ap + 5, att * A1.y);
  atomAddF(ap + 6, att * A1.z); atomAddF(ap + 7, att * A1.w);
}

// ---- finalize: scatter-mean + ReLU ----------------------------------------
__global__ void k_finalize(const float* __restrict__ accum, const int* __restrict__ cnt,
                           float* __restrict__ out, int total) {
  int i = blockIdx.x * blockDim.x + threadIdx.x;
  if (i >= total) return;
  float v = accum[i] / (float)cnt[i >> 6];
  out[i] = v > 0.f ? v : 0.f;
}

// ---------------------------------------------------------------------------
extern "C" void kernel_launch(void* const* d_in, const int* in_sizes, int n_in,
                              void* d_out, int out_size, void* d_ws, size_t ws_size,
                              hipStream_t stream) {
  const float* x    = (const float*)d_in[0];
  const float* W    = (const float*)d_in[1];
  const float* qry  = (const float*)d_in[2];
  const int*   ni   = (const int*)d_in[3];
  const int*   no   = (const int*)d_in[4];
  const int*   rel  = (const int*)d_in[5];
  const float* ew   = (const float*)d_in[6];
  float* out = (float*)d_out;

  const int N = in_sizes[0] / 64;
  const int E = in_sizes[3];
  const int M = E + N;           // edges + self-loops

  // workspace layout (256B-aligned slabs)
  char* ws = (char*)d_ws;
  size_t off = 0;
  auto take = [&](size_t bytes) {
    char* p = ws + off;
    off = (off + bytes + 255) & ~(size_t)255;
    return p;
  };
  __bf16*  xh     = (__bf16*)take((size_t)N * 64 * 2);
  __bf16*  Wh     = (__bf16*)take((size_t)RP1 * 64 * 64 * 2);
  float*   hidden = (float*)take((size_t)RP1 * N * 64 * 4);   // 115 MB, L2-resident
  float*   wbuf   = (float*)take((size_t)M * NHEAD * 4);
  char*    zbase  = ws + off;                                  // zero-init region start
  unsigned* mxkey = (unsigned*)take((size_t)N * NHEAD * 4);
  float*    ssum  = (float*)take((size_t)N * NHEAD * 4);
  int*      cnt   = (int*)take((size_t)N * 4);
  float*    accum = (float*)take((size_t)N * 64 * 4);
  int zn = (int)(((ws + off) - zbase) / 4);

  const int TB = 256;
  int nprep = N * 64;
  k_prep_bf16<<<(nprep + TB - 1) / TB, TB, 0, stream>>>(x, W, xh, Wh, N * 64, RP1 * 64 * 64);

  k_zero_u32<<<(zn + TB - 1) / TB, TB, 0, stream>>>((unsigned*)zbase, zn);

  int ntask   = RP1 * (N >> 4);           // waves of 16-row strips
  int nblocks = (ntask + 7) / 8;          // 8 waves per 256-thread block
  k_hidden_wmma<<<nblocks, TB, 0, stream>>>(xh, Wh, hidden, N);

  int nedge = M * NHEAD;
  k_edge_logits<<<(nedge + TB - 1) / TB, TB, 0, stream>>>(hidden, qry, ni, no, rel,
                                                          wbuf, mxkey, cnt, N, E, M);
  k_edge_exp<<<(nedge + TB - 1) / TB, TB, 0, stream>>>(no, ew, wbuf, mxkey, ssum, E, M);
  k_edge_scatter<<<(nedge + TB - 1) / TB, TB, 0, stream>>>(hidden, ni, no, rel,
                                                           wbuf, ssum, cnt, accum, N, E, M);
  k_finalize<<<(out_size + TB - 1) / TB, TB, 0, stream>>>(accum, cnt, out, out_size);
}
